// MambaEncoder_47278999994441
// MI455X (gfx1250) — matrix operand order repaired and checked by reading, not compiled, not run on hardware
//
#include <hip/hip_runtime.h>
#include <hip/hip_bf16.h>
#include <math.h>

typedef _Float16 h16;
typedef __attribute__((ext_vector_type(16))) _Float16 v16h;
typedef __attribute__((ext_vector_type(8)))  _Float16 v8h;
typedef __attribute__((ext_vector_type(8)))  float    v8f;

#define BB 2048
#define LL 256
#define MCH 32
#define DI 64
#define DSTATE 16
#define NLAYER 2
#define NROWS (BB*LL)   // 524288

// ---------------- workspace layout (bytes) ----------------
static const size_t OFF_SEQ   = 0;            // f32 [NROWS x 32]     = 67,108,864
static const size_t OFF_XZ    = 67108864;     // f16 [NROWS x 128]    = 134,217,728
static const size_t OFF_XC    = 201326592;    // f16 [NROWS x 64]     = 67,108,864 (reused as ysg)
static const size_t OFF_DBL   = 268435456;    // f32 [NROWS x 34]     = 71,303,168
static const size_t OFF_STATS = 339738624;    // f32 [NROWS x 2]      = 4,194,304
static const size_t OFF_FCIN  = 343932928;    // f16 [2048 x 2048]    = 8,388,608
static const size_t OFF_FCRAW = 352321536;    // f32 [2048 x 512]     = 4,194,304
static const size_t OFF_STEMQ = 356515840;    // f32 576
static const size_t OFF_PROJQ = 356519936;    // f32 256
static const size_t OFF_FCWT  = 356524032;    // f16 packed fc weight fragments (2 MB)
static const size_t OFF_WIN   = 358621184;    // f16 packed, 2*4096
static const size_t OFF_WXP   = 358637568;    // f16 packed, 2*3072
static const size_t OFF_WOUT  = 358653952;    // f16 packed, 2*2048
static const size_t OFF_CELLS = 358670336;    // u32 [8] min/max cells

// ---------------- helpers ----------------
static __device__ __forceinline__ unsigned fkey(float f) {
  unsigned u = __float_as_uint(f);
  return (u & 0x80000000u) ? ~u : (u | 0x80000000u);   // monotonic float->uint
}
static __device__ __forceinline__ float funkey(unsigned k) {
  unsigned u = (k & 0x80000000u) ? (k ^ 0x80000000u) : ~k;
  return __uint_as_float(u);
}
static __device__ __forceinline__ unsigned umin_(unsigned a, unsigned b){ return a<b?a:b; }
static __device__ __forceinline__ unsigned umax_(unsigned a, unsigned b){ return a>b?a:b; }
static __device__ __forceinline__ float siluf(float x){ return x / (1.f + __expf(-x)); }

static __device__ __forceinline__ v8f wmma16(v16h a, v16h b, v8f c) {
  // D = A(16x32 f16) * B(32x16 f16) + C(f32), wave32
  return __builtin_amdgcn_wmma_f32_16x16x32_f16(false, a, false, b, (short)0, c, false, false);
}
// 16 contiguous halves (32 B) -> one fragment register set, two b128 loads
static __device__ __forceinline__ v16h ldfrag32B(const h16* p) {
  v8h lo = *(const v8h*)p;
  v8h hi = *(const v8h*)(p + 8);
  return __builtin_shufflevector(lo, hi, 0,1,2,3,4,5,6,7,8,9,10,11,12,13,14,15);
}
// A fragment: two contiguous 16B runs at k0+8*hi and k0+16+8*hi
static __device__ __forceinline__ v16h ldfragA(const h16* arow, int k0, int hi) {
  v8h lo = *(const v8h*)(arow + k0 + 8*hi);
  v8h hh = *(const v8h*)(arow + k0 + 16 + 8*hi);
  return __builtin_shufflevector(lo, hh, 0,1,2,3,4,5,6,7,8,9,10,11,12,13,14,15);
}

// ---------------- tiny utility kernels ----------------
__global__ void init_cells_kernel(unsigned* cells) {
  int i = threadIdx.x;
  if (i < 8) cells[i] = (i & 1) ? 0u : 0xFFFFFFFFu;
}

__global__ void minmax_kernel(const float* __restrict__ w, int n, unsigned* cells) {
  __shared__ unsigned smn[256], smx[256];
  unsigned mn = 0xFFFFFFFFu, mx = 0u;
  for (int i = blockIdx.x*blockDim.x + threadIdx.x; i < n; i += gridDim.x*blockDim.x) {
    unsigned k = fkey(w[i]);
    mn = umin_(mn, k); mx = umax_(mx, k);
  }
  smn[threadIdx.x] = mn; smx[threadIdx.x] = mx;
  __syncthreads();
  for (int s = 128; s > 0; s >>= 1) {
    if ((int)threadIdx.x < s) {
      smn[threadIdx.x] = umin_(smn[threadIdx.x], smn[threadIdx.x+s]);
      smx[threadIdx.x] = umax_(smx[threadIdx.x], smx[threadIdx.x+s]);
    }
    __syncthreads();
  }
  if (threadIdx.x == 0) { atomicMin(&cells[0], smn[0]); atomicMax(&cells[1], smx[0]); }
}

static __device__ __forceinline__ float fquant(float t, float mn, float mx) {
  float step = (mx - mn) * (1.f/255.f);
  if (step == 0.f) return t;
  float c = fminf(fmaxf(t, mn), mx);
  return rintf((c - mn)/step)*step + mn;          // rintf = round-to-nearest-even
}

__global__ void quantw_kernel(const float* __restrict__ w, int n,
                              const unsigned* __restrict__ cells, float* __restrict__ out) {
  int i = blockIdx.x*blockDim.x + threadIdx.x;
  if (i >= n) return;
  out[i] = fquant(w[i], funkey(cells[0]), funkey(cells[1]));
}

// Pack a K x N (row-major, ldb=N) f32 weight into WMMA B-fragment-major f16:
// dst[(((ks*ntiles + tn)*32 + lane)*16 + e] = B[ks*32 + 8*(lane>>4) + (e&7) + 16*(e>>3)][tn*16 + (lane&15)]
__global__ void packB_kernel(const float* __restrict__ Bsrc, int N, int K,
                             h16* __restrict__ dst) {
  int ntiles = (N + 15) >> 4;
  int total = (K >> 5) * ntiles * 512;
  int idx = blockIdx.x*blockDim.x + threadIdx.x;
  if (idx >= total) return;
  int e    = idx & 15;
  int lane = (idx >> 4) & 31;
  int tn   = (idx >> 9) % ntiles;
  int ks   = (idx >> 9) / ntiles;
  int kb   = ks*32 + ((lane >> 4) << 3) + (e & 7) + ((e >> 3) << 4);
  int col  = tn*16 + (lane & 15);
  float v = (col < N) ? Bsrc[(size_t)kb*N + col] : 0.f;
  dst[idx] = (h16)v;
}

// fake-quant fc_w (512 x 2048) and store as packed B fragments of B = fc_w^T (2048 x 512)
__global__ void quantfc_pack_kernel(const float* __restrict__ fw,
                                    const unsigned* __restrict__ cells, h16* __restrict__ dst) {
  int idx = blockIdx.x*blockDim.x + threadIdx.x;     // 2048*512 packed elements
  if (idx >= 2048*512) return;
  int e    = idx & 15;
  int lane = (idx >> 4) & 31;
  int tn   = (idx >> 9) & 31;      // ntiles = 32
  int ks   = idx >> 14;            // 0..63
  int kb   = ks*32 + ((lane >> 4) << 3) + (e & 7) + ((e >> 3) << 4);       // 0..2047
  int col  = tn*16 + (lane & 15);                  // 0..511
  float q = fquant(fw[(size_t)col*2048 + kb], funkey(cells[0]), funkey(cells[1]));
  dst[idx] = (h16)q;
}

// ---------------- stem conv + BN + SiLU -> seq (b, l, m) ----------------
__global__ void stem_kernel(const float* __restrict__ x, const float* __restrict__ wq,
                            const float* __restrict__ gam, const float* __restrict__ bet,
                            const float* __restrict__ mean, const float* __restrict__ var,
                            float* __restrict__ seq) {
  int idx = blockIdx.x*blockDim.x + threadIdx.x;       // (b*256+l)*32+m
  if (idx >= BB*LL*MCH) return;
  int m = idx & 31;
  int l = (idx >> 5) & 255;
  int b = idx >> 13;
  int oh = l >> 4, ow = l & 15;
  float s = 0.f;
  #pragma unroll
  for (int ci = 0; ci < 2; ++ci)
    #pragma unroll
    for (int kh = 0; kh < 3; ++kh) {
      int ih = oh*2 - 1 + kh;
      if ((unsigned)ih >= 32u) continue;
      #pragma unroll
      for (int kw = 0; kw < 3; ++kw) {
        int iw = ow*2 - 1 + kw;
        if ((unsigned)iw >= 32u) continue;
        s += x[(((size_t)b*2 + ci)*32 + ih)*32 + iw] * wq[((m*2 + ci)*3 + kh)*3 + kw];
      }
    }
  s = (s - mean[m]) * rsqrtf(var[m] + 1e-5f) * gam[m] + bet[m];
  seq[idx] = siluf(s);
}

// ---------------- layernorm statistics (mu, rsqrt(var+eps)) per row ----------------
__global__ void ln_stats_kernel(const float* __restrict__ seq, float* __restrict__ stats, int nrows) {
  int r = blockIdx.x*blockDim.x + threadIdx.x;
  if (r >= nrows) return;
  const float* p = seq + (size_t)r*MCH;
  float s = 0.f;
  #pragma unroll
  for (int i = 0; i < MCH; ++i) s += p[i];
  float mu = s * (1.f/MCH);
  float v = 0.f;
  #pragma unroll
  for (int i = 0; i < MCH; ++i) { float d = p[i]-mu; v += d*d; }
  stats[2*(size_t)r]   = mu;
  stats[2*(size_t)r+1] = rsqrtf(v*(1.f/MCH) + 1e-5f);
}

// ---------------- LN-fused WMMA GEMM: (NROWS x 32) @ (32 x 128) -> xz f16 ----------------
// block(32,4): each wave does 2 N-tiles (2 WMMAs), LN applied while building A fragment.
__global__ void gemm_ln_wmma_kernel(const float* __restrict__ X, const float* __restrict__ stats,
                                    const float* __restrict__ g, const float* __restrict__ bet,
                                    const h16* __restrict__ Bp, h16* __restrict__ Co16) {
  int lane = threadIdx.x;
  int tn0 = threadIdx.y * 2;                           // 0,2,4,6
  int tm = blockIdx.y;
  int row0 = tm*16;
  int n15 = lane & 15, hi = lane >> 4;
  int row = row0 + n15;
  float mu = stats[2*(size_t)row], rs = stats[2*(size_t)row + 1];

  const float* Xr = X + (size_t)row*MCH;
  float4 x0 = *(const float4*)(Xr + 8*hi);
  float4 x1 = *(const float4*)(Xr + 8*hi + 4);
  float4 x2 = *(const float4*)(Xr + 16 + 8*hi);
  float4 x3 = *(const float4*)(Xr + 16 + 8*hi + 4);
  float4 g0 = *(const float4*)(g + 8*hi);
  float4 g1 = *(const float4*)(g + 8*hi + 4);
  float4 g2 = *(const float4*)(g + 16 + 8*hi);
  float4 g3 = *(const float4*)(g + 16 + 8*hi + 4);
  float4 b0v = *(const float4*)(bet + 8*hi);
  float4 b1v = *(const float4*)(bet + 8*hi + 4);
  float4 b2v = *(const float4*)(bet + 16 + 8*hi);
  float4 b3v = *(const float4*)(bet + 16 + 8*hi + 4);
  float xs[16] = {x0.x,x0.y,x0.z,x0.w, x1.x,x1.y,x1.z,x1.w,
                  x2.x,x2.y,x2.z,x2.w, x3.x,x3.y,x3.z,x3.w};
  float gs[16] = {g0.x,g0.y,g0.z,g0.w, g1.x,g1.y,g1.z,g1.w,
                  g2.x,g2.y,g2.z,g2.w, g3.x,g3.y,g3.z,g3.w};
  float bs[16] = {b0v.x,b0v.y,b0v.z,b0v.w, b1v.x,b1v.y,b1v.z,b1v.w,
                  b2v.x,b2v.y,b2v.z,b2v.w, b3v.x,b3v.y,b3v.z,b3v.w};
  v16h a;
  #pragma unroll
  for (int i = 0; i < 16; ++i) a[i] = (h16)((xs[i] - mu)*rs*gs[i] + bs[i]);

  v16h bf0 = ldfrag32B(Bp + ((size_t)(tn0    )*32 + lane)*16);
  v16h bf1 = ldfrag32B(Bp + ((size_t)(tn0 + 1)*32 + lane)*16);
  v8f acc0 = {}, acc1 = {};
  acc0 = wmma16(a, bf0, acc0);
  acc1 = wmma16(a, bf1, acc1);

  int col0 = tn0*16 + n15;
  #pragma unroll
  for (int v = 0; v < 8; ++v) {
    int crow = row0 + v + 8*hi;                        // C/D layout: M = v + 8*hi
    Co16[(size_t)crow*128 + col0]      = (h16)acc0[v];
    Co16[(size_t)crow*128 + col0 + 16] = (h16)acc1[v];
  }
}

// ---------------- generic WMMA GEMM: C = A(f16) @ Bpacked(f16) [+ Cin(f32)] ----------------
// Each wave: one 16-row block x two 16-col tiles, A fragment reused across both WMMAs.
__global__ void gemm_wmma_kernel(const h16* __restrict__ A, int lda,
                                 const h16* __restrict__ Bp, int ntiles,
                                 const float* __restrict__ Cin,
                                 float* __restrict__ Co32, h16* __restrict__ Co16,
                                 int ldc, int N, int K) {
  int lane = threadIdx.x;
  int tn0 = (blockIdx.x*blockDim.y + threadIdx.y)*2;
  if (tn0 >= ntiles) return;
  bool t1 = (tn0 + 1) < ntiles;
  int tm = blockIdx.y;
  int row0 = tm*16;
  int n15 = lane & 15, hi = lane >> 4;
  int row = row0 + n15;
  int col0 = tn0*16 + n15;
  int col1 = col0 + 16;

  v8f acc0, acc1;
  #pragma unroll
  for (int v = 0; v < 8; ++v) {
    int crow = row0 + v + 8*hi;
    acc0[v] = (Cin && col0 < N)       ? Cin[(size_t)crow*ldc + col0] : 0.f;
    acc1[v] = (Cin && t1 && col1 < N) ? Cin[(size_t)crow*ldc + col1] : 0.f;
  }
  const h16* Arow = A + (size_t)row*lda;
  int ksteps = K >> 5;
  for (int ks = 0; ks < ksteps; ++ks) {
    v16h a = ldfragA(Arow, ks*32, hi);
    v16h bf0 = ldfrag32B(Bp + (((size_t)ks*ntiles + tn0)*32 + lane)*16);
    acc0 = wmma16(a, bf0, acc0);
    if (t1) {
      v16h bf1 = ldfrag32B(Bp + (((size_t)ks*ntiles + tn0 + 1)*32 + lane)*16);
      acc1 = wmma16(a, bf1, acc1);
    }
  }
  #pragma unroll
  for (int v = 0; v < 8; ++v) {
    int crow = row0 + v + 8*hi;
    if (col0 < N) {
      if (Co32) Co32[(size_t)crow*ldc + col0] = acc0[v];
      if (Co16) Co16[(size_t)crow*ldc + col0] = (h16)acc0[v];
    }
    if (t1 && col1 < N) {
      if (Co32) Co32[(size_t)crow*ldc + col1] = acc1[v];
      if (Co16) Co16[(size_t)crow*ldc + col1] = (h16)acc1[v];
    }
  }
}

// ---------------- depthwise causal conv1d (k=4) + SiLU ----------------
__global__ void conv1d_kernel(const h16* __restrict__ xz, const float* __restrict__ cw,
                              const float* __restrict__ cb, h16* __restrict__ xc) {
  int idx = blockIdx.x*blockDim.x + threadIdx.x;       // (b*256+l)*64+d
  if (idx >= NROWS*DI) return;
  int d = idx & 63;
  int l = (idx >> 6) & 255;
  size_t bl = (size_t)(idx >> 6);
  size_t base = (bl - l) * 128;                        // b*256*128
  float s = cb[d];
  #pragma unroll
  for (int k = 0; k < 4; ++k) {
    int ll = l - 3 + k;
    if (ll >= 0) s += cw[d*4 + k] * (float)xz[base + (size_t)ll*128 + d];
  }
  xc[idx] = (h16)siluf(s);
}

// ---------------- selective scan, fused softplus(dt) + D-skip + SiLU(z) gate ----------------
__global__ void scan_kernel(const float* __restrict__ dbl, const h16* __restrict__ xz,
                            h16* __restrict__ xc,
                            const float* __restrict__ Wdt, const float* __restrict__ dtb,
                            const float* __restrict__ Alog, const float* __restrict__ Dssm) {
  __shared__ float sh[4][34];
  int t = threadIdx.x;
  int sub = t >> 6, d = t & 63;
  int b = blockIdx.x*4 + sub;
  float w0 = Wdt[d], w1 = Wdt[64 + d], bias = dtb[d], Dd = Dssm[d];
  float An[DSTATE], hs[DSTATE];
  #pragma unroll
  for (int s = 0; s < DSTATE; ++s) { An[s] = -__expf(Alog[d*DSTATE + s]); hs[s] = 0.f; }
  for (int l = 0; l < LL; ++l) {
    size_t bl = (size_t)b*LL + l;
    if (d < 34) sh[sub][d] = dbl[bl*34 + d];
    if (l + 1 < LL) __builtin_prefetch(&dbl[(bl + 1)*34], 0, 1);
    __syncthreads();
    float dtv = sh[sub][0]*w0 + sh[sub][1]*w1 + bias;
    dtv = (dtv > 20.f) ? dtv : log1pf(__expf(dtv));    // softplus
    float xv = (float)xc[bl*DI + d];
    float dtx = dtv * xv;
    float y = 0.f;
    #pragma unroll
    for (int s = 0; s < DSTATE; ++s) {
      float dA = __expf(dtv * An[s]);
      hs[s] = hs[s]*dA + dtx*sh[sub][2 + s];
      y += hs[s]*sh[sub][18 + s];
    }
    float zv = (float)xz[bl*128 + DI + d];
    float out = (y + xv*Dd) * siluf(zv);
    __syncthreads();
    xc[bl*DI + d] = (h16)out;                          // in-place: xc read above first
  }
}

// ---------------- 1x1 proj conv -> flattened fc input (f16) ----------------
__global__ void proj_kernel(const float* __restrict__ seq, const float* __restrict__ pw,
                            h16* __restrict__ fcin) {
  int idx = blockIdx.x*blockDim.x + threadIdx.x;       // b*2048 + o*256 + l
  if (idx >= BB*8*LL) return;
  int l = idx & 255;
  int o = (idx >> 8) & 7;
  int b = idx >> 11;
  const float* row = seq + ((size_t)b*LL + l)*MCH;
  float s = 0.f;
  #pragma unroll
  for (int m = 0; m < MCH; ++m) s += row[m]*pw[o*MCH + m];
  fcin[idx] = (h16)s;
}

// ---------------- fc epilogue: +bias, sigmoid, global min/max of result ----------------
__global__ void fcepi_kernel(float* __restrict__ fcraw, const float* __restrict__ fcb,
                             unsigned* cells) {
  __shared__ unsigned smn[256], smx[256];
  int i = blockIdx.x*256 + threadIdx.x;                // exactly BB*512 threads
  float v = fcraw[i] + fcb[i & 511];
  v = 1.f / (1.f + __expf(-v));
  fcraw[i] = v;
  unsigned k = fkey(v);
  smn[threadIdx.x] = k; smx[threadIdx.x] = k;
  __syncthreads();
  for (int s = 128; s > 0; s >>= 1) {
    if ((int)threadIdx.x < s) {
      smn[threadIdx.x] = umin_(smn[threadIdx.x], smn[threadIdx.x+s]);
      smx[threadIdx.x] = umax_(smx[threadIdx.x], smx[threadIdx.x+s]);
    }
    __syncthreads();
  }
  if (threadIdx.x == 0) { atomicMin(&cells[0], smn[0]); atomicMax(&cells[1], smx[0]); }
}

__global__ void outq_kernel(const float* __restrict__ fcraw, const unsigned* __restrict__ cells,
                            float* __restrict__ out) {
  int i = blockIdx.x*blockDim.x + threadIdx.x;
  if (i >= BB*512) return;
  out[i] = fquant(fcraw[i], funkey(cells[0]), funkey(cells[1]));
}

// ---------------- host launch ----------------
extern "C" void kernel_launch(void* const* d_in, const int* in_sizes, int n_in,
                              void* d_out, int out_size, void* d_ws, size_t ws_size,
                              hipStream_t stream) {
  (void)in_sizes; (void)n_in; (void)out_size; (void)ws_size;
  const float* x      = (const float*)d_in[0];
  const float* stem_w = (const float*)d_in[1];
  const float* bn_g   = (const float*)d_in[2];
  const float* bn_b   = (const float*)d_in[3];
  const float* bn_m   = (const float*)d_in[4];
  const float* bn_v   = (const float*)d_in[5];
  const float* ln_g   = (const float*)d_in[6];
  const float* ln_b   = (const float*)d_in[7];
  const float* W_in   = (const float*)d_in[8];
  const float* conv_w = (const float*)d_in[9];
  const float* conv_b = (const float*)d_in[10];
  const float* W_xp   = (const float*)d_in[11];
  const float* W_dt   = (const float*)d_in[12];
  const float* dt_b   = (const float*)d_in[13];
  const float* A_log  = (const float*)d_in[14];
  const float* D_ssm  = (const float*)d_in[15];
  const float* W_out  = (const float*)d_in[16];
  const float* proj_w = (const float*)d_in[17];
  const float* fc_w   = (const float*)d_in[18];
  const float* fc_b   = (const float*)d_in[19];

  char* ws = (char*)d_ws;
  float*    seq   = (float*)(ws + OFF_SEQ);
  h16*      xz    = (h16*)(ws + OFF_XZ);
  h16*      xc    = (h16*)(ws + OFF_XC);
  float*    dbl   = (float*)(ws + OFF_DBL);
  float*    stats = (float*)(ws + OFF_STATS);
  h16*      fcin  = (h16*)(ws + OFF_FCIN);
  float*    fcraw = (float*)(ws + OFF_FCRAW);
  float*    stemq = (float*)(ws + OFF_STEMQ);
  float*    projq = (float*)(ws + OFF_PROJQ);
  h16*      fcwp  = (h16*)(ws + OFF_FCWT);
  h16*      winp  = (h16*)(ws + OFF_WIN);
  h16*      wxpp  = (h16*)(ws + OFF_WXP);
  h16*      woutp = (h16*)(ws + OFF_WOUT);
  unsigned* cells = (unsigned*)(ws + OFF_CELLS);

  // weight fake-quant + packed f16 weight staging
  init_cells_kernel<<<1, 32, 0, stream>>>(cells);
  minmax_kernel<<<4,   256, 0, stream>>>(stem_w, 576,      cells + 0);
  minmax_kernel<<<2,   256, 0, stream>>>(proj_w, 256,      cells + 2);
  minmax_kernel<<<256, 256, 0, stream>>>(fc_w,   512*2048, cells + 4);
  quantw_kernel<<<3, 256, 0, stream>>>(stem_w, 576, cells + 0, stemq);
  quantw_kernel<<<1, 256, 0, stream>>>(proj_w, 256, cells + 2, projq);
  quantfc_pack_kernel<<<4096, 256, 0, stream>>>(fc_w, cells + 4, fcwp);
  for (int layer = 0; layer < NLAYER; ++layer) {
    packB_kernel<<<16, 256, 0, stream>>>(W_in  + (size_t)layer*MCH*128, 128, MCH, winp  + (size_t)layer*4096);
    packB_kernel<<<12, 256, 0, stream>>>(W_xp  + (size_t)layer*DI*34,   34,  DI,  wxpp  + (size_t)layer*3072);
    packB_kernel<<<8,  256, 0, stream>>>(W_out + (size_t)layer*DI*MCH,  MCH, DI,  woutp + (size_t)layer*2048);
  }

  // stem conv + BN + SiLU, written directly in (b, l, m) sequence layout
  stem_kernel<<<(BB*LL*MCH + 255)/256, 256, 0, stream>>>(x, stemq, bn_g, bn_b, bn_m, bn_v, seq);

  for (int layer = 0; layer < NLAYER; ++layer) {
    ln_stats_kernel<<<(NROWS + 255)/256, 256, 0, stream>>>(seq, stats, NROWS);
    // xz = LN(seq) @ W_in   (524288x32 @ 32x128), LN fused into A-fragment build
    gemm_ln_wmma_kernel<<<dim3(1, NROWS/16), dim3(32, 4), 0, stream>>>(
        seq, stats, ln_g + layer*MCH, ln_b + layer*MCH, winp + (size_t)layer*4096, xz);
    // depthwise causal conv (k=4) + SiLU on the xs half
    conv1d_kernel<<<(NROWS*DI + 255)/256, 256, 0, stream>>>(
        xz, conv_w + layer*DI*4, conv_b + layer*DI, xc);
    // dbl = xc @ W_xproj    (524288x64 @ 64x34), 3 N-tiles -> 2 waves/row-block
    gemm_wmma_kernel<<<dim3(1, NROWS/16), dim3(32, 2), 0, stream>>>(
        xc, DI, wxpp + (size_t)layer*3072, 3, nullptr, dbl, nullptr, 34, 34, DI);
    // selective scan (fused softplus / gate / D-skip), ysg written over xc
    scan_kernel<<<BB/4, 256, 0, stream>>>(
        dbl, xz, xc, W_dt + layer*2*DI, dt_b + layer*DI,
        A_log + layer*DI*DSTATE, D_ssm + layer*DI);
    // seq += ysg @ W_out    (524288x64 @ 64x32), residual via Cin=seq, 1 wave/row-block
    gemm_wmma_kernel<<<dim3(1, NROWS/16), dim3(32, 1), 0, stream>>>(
        xc, DI, woutp + (size_t)layer*2048, 2, seq, seq, nullptr, MCH, MCH, DI);
  }

  // 1x1 proj conv, flattened into fc input rows (f16)
  proj_kernel<<<(BB*8*LL + 255)/256, 256, 0, stream>>>(seq, projq, fcin);
  // fc: (2048x2048) @ (2048x512), 32 N-tiles -> 16 tile-pairs, 8 waves/block
  gemm_wmma_kernel<<<dim3(2, BB/16), dim3(32, 8), 0, stream>>>(
      fcin, 2048, fcwp, 32, nullptr, fcraw, nullptr, 512, 512, 2048);
  // bias + sigmoid + global min/max, then 8-bit fake quant to d_out
  fcepi_kernel<<<(BB*512)/256, 256, 0, stream>>>(fcraw, fc_b, cells + 6);
  outq_kernel<<<(BB*512 + 255)/256, 256, 0, stream>>>(fcraw, cells + 6, (float*)d_out);
}
